// SimpleModel_37658273251597
// MI455X (gfx1250) — compile-verified, matrix operand order checked by reading, not run on hardware
//
#include <hip/hip_runtime.h>
#include <cstdint>
#include <cstddef>

typedef unsigned short u16;
typedef __attribute__((ext_vector_type(16))) __bf16 v16bf;
typedef __attribute__((ext_vector_type(8)))  float  v8f;

struct U32x8  { uint4 a, b; };
struct U16x16 { u16 v[16]; };

// ---------- helpers ----------
__device__ __forceinline__ u16 f2bf(float f) {
  unsigned u = __builtin_bit_cast(unsigned, f);
  u += 0x7fffu + ((u >> 16) & 1u);         // round-to-nearest-even
  return (u16)(u >> 16);
}

// Lane fragment for 16-bit WMMA operands (ISA 7.12.2): the lane's 16
// halfwords sit at contraction indices [hi*8..hi*8+7] and [16+hi*8..+7]
// (hi = lane>>4) -> two contiguous 16-byte chunks -> 2x b128 loads.
__device__ __forceinline__ v16bf load_frag2(const u16* p, int hi) {
  U32x8 t;
  t.a = *(const uint4*)(p + hi * 8);
  t.b = *(const uint4*)(p + 16 + hi * 8);
  return __builtin_bit_cast(v16bf, t);
}

// ---------- weight conversion ----------
__global__ void cvt_bf16_kernel(const float* __restrict__ src, u16* __restrict__ dst, int n) {
  for (int i = blockIdx.x * blockDim.x + threadIdx.x; i < n; i += gridDim.x * blockDim.x)
    dst[i] = f2bf(src[i]);
}

// [E,V] fp32 -> [E,Vp] bf16 (padded leading dim so GEMM staging is unguarded)
__global__ void cvt_pad_kernel(const float* __restrict__ src, u16* __restrict__ dst,
                               int E_, int V_, int Vp_) {
  int n = E_ * V_;
  for (int i = blockIdx.x * blockDim.x + threadIdx.x; i < n; i += gridDim.x * blockDim.x) {
    int v = i % V_, e = i / V_;
    dst[(size_t)e * Vp_ + v] = f2bf(src[i]);
  }
}

// [L,NH,E,HS] fp32 -> [L, E, NH*HS] bf16 (QKV = one GEMM per matrix per layer)
__global__ void cvt_qkv_kernel(const float* __restrict__ src, u16* __restrict__ dst,
                               int L_, int NH_, int E_, int HS_) {
  int n = L_ * NH_ * E_ * HS_;
  for (int i = blockIdx.x * blockDim.x + threadIdx.x; i < n; i += gridDim.x * blockDim.x) {
    int d = i % HS_; int t = i / HS_;
    int e = t % E_;  t /= E_;
    int h = t % NH_; int l = t / NH_;
    dst[((size_t)(l * E_ + e)) * (NH_ * HS_) + h * HS_ + d] = f2bf(src[i]);
  }
}

// ---------- embedding ----------
__global__ void embed_kernel(const int* __restrict__ idx, const float* __restrict__ tok,
                             const float* __restrict__ pos, u16* __restrict__ xb,
                             int T_, int E_) {
  int row = blockIdx.x;          // b*T + t
  int e   = threadIdx.x;         // blockDim.x == E
  int t   = row % T_;
  int tk  = idx[row];
  xb[(size_t)row * E_ + e] = f2bf(tok[(size_t)tk * E_ + e] + pos[(size_t)t * E_ + e]);
}

// ---------- WMMA GEMM: C[M,N] = A[M,K](bf16) * B[K,N](bf16) ----------
// 128 threads = 4 waves; block tile 16x128; each wave two 16x16 C tiles.
// B tile (32 x 128) staged TRANSPOSED in LDS with unguarded aligned b128 loads
// (caller guarantees ldb covers blockIdx.x*128+128 columns; pad region harmless).
// Compile-time epilogue variants: bias add, ReLU, fp32-vs-bf16 out, transposed out.
template<bool HAS_BIAS, bool RELU, bool OUT_F32, bool TRANSC>
__global__ __launch_bounds__(128)
void gemm_bf16_kernel(const u16* __restrict__ A, int lda,
                      const u16* __restrict__ Bm, int ldb,
                      float* __restrict__ Cf, u16* __restrict__ Cb,
                      const float* __restrict__ bias,
                      int M, int N, int K) {
  __shared__ alignas(16) u16 Bt[128][32];     // [n_local][k_local]
  const int tid  = threadIdx.x;
  const int lane = tid & 31;
  const int wave = tid >> 5;
  const int hi   = lane >> 4, ln = lane & 15;
  const int tnb  = blockIdx.x * 128;          // block column base
  const int rb   = blockIdx.y * 16;           // block row base
  const int arow = rb + ln;                   // A-frag row for this lane

  const u16* Ap = A + (size_t)arow * lda;
  v8f acc[2];
  acc[0] = (v8f){0.f,0.f,0.f,0.f,0.f,0.f,0.f,0.f};
  acc[1] = (v8f){0.f,0.f,0.f,0.f,0.f,0.f,0.f,0.f};

  // staging coords: each thread moves 32 consecutive n for one k row
  const int kk = (tid * 32) >> 7;             // 0..31
  const int n0 = (tid * 32) & 127;

  for (int k0 = 0; k0 < K; k0 += 32) {
    {
      const uint4* p4 = (const uint4*)(Bm + (size_t)(k0 + kk) * ldb + tnb + n0);
#pragma unroll
      for (int j = 0; j < 2; j++) {
        U32x8 ch; ch.a = p4[2 * j]; ch.b = p4[2 * j + 1];
        U16x16 c16 = __builtin_bit_cast(U16x16, ch);
#pragma unroll
        for (int i = 0; i < 16; i++)
          Bt[n0 + j * 16 + i][kk] = c16.v[i];
      }
    }
    __syncthreads();
    __builtin_prefetch(Ap + k0 + 64, 0, 1);   // global_prefetch next A chunk
    v16bf af = load_frag2(Ap + k0, hi);
#pragma unroll
    for (int t = 0; t < 2; t++) {
      v16bf bf = load_frag2(&Bt[wave * 32 + t * 16 + ln][0], hi);
      acc[t] = __builtin_amdgcn_wmma_f32_16x16x32_bf16(false, af, false, bf,
                                                       (short)0, acc[t], false, false);
    }
    __syncthreads();
  }

  // Epilogue. C/D layout: vgpr r -> row rb + r + hi*8 ; lane&15 -> column.
  // All per-column work hoisted out of the row loop.
#pragma unroll
  for (int t = 0; t < 2; t++) {
    const int col = tnb + wave * 32 + t * 16 + ln;
    if (col < N) {
      float bv = HAS_BIAS ? bias[col] : 0.f;
      if (OUT_F32) {
        float* out = Cf + (size_t)(rb + hi * 8) * N + col;
#pragma unroll
        for (int r = 0; r < 8; r++) {
          float v = acc[t][r] + bv;
          if (RELU) v = fmaxf(v, 0.f);
          out[(size_t)r * N] = v;
        }
      } else if (TRANSC) {
        u16* out = Cb + (size_t)col * M + rb + hi * 8;
#pragma unroll
        for (int r = 0; r < 8; r++) {
          float v = acc[t][r] + bv;
          if (RELU) v = fmaxf(v, 0.f);
          out[r] = f2bf(v);
        }
      } else {
        u16* out = Cb + (size_t)(rb + hi * 8) * N + col;
#pragma unroll
        for (int r = 0; r < 8; r++) {
          float v = acc[t][r] + bv;
          if (RELU) v = fmaxf(v, 0.f);
          out[(size_t)r * N] = f2bf(v);
        }
      }
    }
  }
}

// ---------- flash attention (one wave32 per 16-query tile per head) ----------
// q,k stored [B*T, E] bf16 head-major columns (col = hd*64 + d);
// v stored TRANSPOSED vt[E][B*T] so the PV B-frag is 2x b128 loads.
__global__ __launch_bounds__(32)
void attn_kernel(const u16* __restrict__ qb, const u16* __restrict__ kb,
                 const u16* __restrict__ vt, u16* __restrict__ xcat,
                 int T_, int Mrows) {
  const int E_ = 256, lane = threadIdx.x;
  const int hi = lane >> 4, ln = lane & 15;
  const int qt = blockIdx.x, hd = blockIdx.y, b = blockIdx.z;
  const int qbase = qt * 16;
  const int hoff  = hd * 64;
  const size_t rb = (size_t)b * T_;

  __shared__ alignas(16) u16 Pl[16][32];      // P tile staging for re-fragmenting

  v16bf qf[2];
#pragma unroll
  for (int ks = 0; ks < 2; ks++)
    qf[ks] = load_frag2(qb + (rb + qbase + ln) * E_ + hoff + ks * 32, hi);

  v8f O[4];
#pragma unroll
  for (int d = 0; d < 4; d++) O[d] = (v8f){0.f,0.f,0.f,0.f,0.f,0.f,0.f,0.f};
  float rmax[8], rsum[8];
#pragma unroll
  for (int r = 0; r < 8; r++) { rmax[r] = -1e30f; rsum[r] = 0.f; }

  for (int s0 = 0; s0 < qbase + 16; s0 += 32) {   // causal: max key = 2047 < T, no clamps
    // S = (q . k^T) * E^-0.5
    v8f S[2];
#pragma unroll
    for (int ns = 0; ns < 2; ns++) {
      v8f c = {0.f,0.f,0.f,0.f,0.f,0.f,0.f,0.f};
      const u16* kp = kb + (rb + s0 + ns * 16 + ln) * E_ + hoff;
#pragma unroll
      for (int ks = 0; ks < 2; ks++) {
        v16bf kf = load_frag2(kp + ks * 32, hi);
        c = __builtin_amdgcn_wmma_f32_16x16x32_bf16(false, qf[ks], false, kf,
                                                    (short)0, c, false, false);
      }
      S[ns] = c;
    }
    // online softmax per row (row m = r + hi*8 lives across one 16-lane half)
#pragma unroll
    for (int r = 0; r < 8; r++) {
      int rowq = qbase + r + hi * 8;
#pragma unroll
      for (int ns = 0; ns < 2; ns++) {
        int key = s0 + ns * 16 + ln;
        float v = S[ns][r] * 0.0625f;                // 1/sqrt(E), E=256
        if (key > rowq) v = -1e30f;                  // causal mask
        S[ns][r] = v;
      }
      float mx = fmaxf(S[0][r], S[1][r]);
#pragma unroll
      for (int off = 1; off < 16; off <<= 1) mx = fmaxf(mx, __shfl_xor(mx, off, 32));
      float nm   = fmaxf(rmax[r], mx);
      float corr = __expf(rmax[r] - nm);
#pragma unroll
      for (int d = 0; d < 4; d++) O[d][r] *= corr;
      float ps = 0.f;
#pragma unroll
      for (int ns = 0; ns < 2; ns++) {
        float p = __expf(S[ns][r] - nm);
        ps += p;
        Pl[r + hi * 8][ns * 16 + ln] = f2bf(p);
      }
#pragma unroll
      for (int off = 1; off < 16; off <<= 1) ps += __shfl_xor(ps, off, 32);
      rsum[r] = rsum[r] * corr + ps;
      rmax[r] = nm;
    }
    __syncthreads();
    // O += P(16x32) . V(32x64); V^T rows are key-contiguous -> 2x b128 frag loads
    v16bf Pf = load_frag2(&Pl[ln][0], hi);
#pragma unroll
    for (int dt = 0; dt < 4; dt++) {
      v16bf Vf = load_frag2(vt + (size_t)(hoff + dt * 16 + ln) * Mrows + rb + s0, hi);
      O[dt] = __builtin_amdgcn_wmma_f32_16x16x32_bf16(false, Pf, false, Vf,
                                                      (short)0, O[dt], false, false);
    }
    __syncthreads();
  }
  // normalize + write concat-head output (bf16 A for the FFN GEMM)
#pragma unroll
  for (int dt = 0; dt < 4; dt++)
#pragma unroll
    for (int r = 0; r < 8; r++) {
      float v = O[dt][r] / rsum[r];
      size_t row = rb + qbase + r + hi * 8;
      xcat[row * E_ + hoff + dt * 16 + ln] = f2bf(v);
    }
}

// ---------- host orchestration ----------
extern "C" void kernel_launch(void* const* d_in, const int* in_sizes, int n_in,
                              void* d_out, int out_size, void* d_ws, size_t ws_size,
                              hipStream_t stream) {
  (void)in_sizes; (void)n_in; (void)out_size; (void)ws_size;
  const int*   idx  = (const int*)  d_in[0];
  const float* tok  = (const float*)d_in[1];
  const float* pos  = (const float*)d_in[2];
  const float* Wq   = (const float*)d_in[3];
  const float* Wk   = (const float*)d_in[4];
  const float* Wv   = (const float*)d_in[5];
  const float* Wff  = (const float*)d_in[6];
  const float* bff  = (const float*)d_in[7];
  const float* Wout = (const float*)d_in[8];
  const float* bout = (const float*)d_in[9];

  constexpr int V = 50257, E = 256, T = 2048, B = 2, L = 3, NH = 4, HS = 64;
  constexpr int Vp = (V + 127) & ~127;         // 50304, multiple of 128
  const int M = B * T;                         // 4096 rows

  char* w = (char*)d_ws;
  auto carve = [&](size_t bytes) -> void* {
    void* p = (void*)w;
    w += (bytes + 255) & ~(size_t)255;
    return p;
  };
  u16* xb     = (u16*)carve((size_t)M * E * 2);
  u16* xcat   = (u16*)carve((size_t)M * E * 2);
  u16* qbuf   = (u16*)carve((size_t)M * E * 2);
  u16* kbuf   = (u16*)carve((size_t)M * E * 2);
  u16* vtbuf  = (u16*)carve((size_t)M * E * 2);    // V transposed [E][M]
  u16* Wq_b   = (u16*)carve((size_t)L * E * E * 2);
  u16* Wk_b   = (u16*)carve((size_t)L * E * E * 2);
  u16* Wv_b   = (u16*)carve((size_t)L * E * E * 2);
  u16* Wff_b  = (u16*)carve((size_t)L * E * E * 2);
  u16* Wout_b = (u16*)carve((size_t)E * Vp * 2);   // padded ldb

  cvt_qkv_kernel<<<256, 256, 0, stream>>>(Wq, Wq_b, L, NH, E, HS);
  cvt_qkv_kernel<<<256, 256, 0, stream>>>(Wk, Wk_b, L, NH, E, HS);
  cvt_qkv_kernel<<<256, 256, 0, stream>>>(Wv, Wv_b, L, NH, E, HS);
  cvt_bf16_kernel<<<256, 256, 0, stream>>>(Wff, Wff_b, L * E * E);
  cvt_pad_kernel<<<2048, 256, 0, stream>>>(Wout, Wout_b, E, V, Vp);

  embed_kernel<<<M, E, 0, stream>>>(idx, tok, pos, xb, T, E);

  const dim3 gProj(E / 128, M / 16);           // 2 x 256 blocks, tile 16x128
  const dim3 gAttn(T / 16, NH, B);
  for (int l = 0; l < L; l++) {
    const size_t wo = (size_t)l * E * E;
    // QKV projections (plain bf16 out; V written transposed)
    gemm_bf16_kernel<false,false,false,false><<<gProj, 128, 0, stream>>>(
        xb, E, Wq_b + wo, E, nullptr, qbuf, nullptr, M, E, E);
    gemm_bf16_kernel<false,false,false,false><<<gProj, 128, 0, stream>>>(
        xb, E, Wk_b + wo, E, nullptr, kbuf, nullptr, M, E, E);
    gemm_bf16_kernel<false,false,false,true><<<gProj, 128, 0, stream>>>(
        xb, E, Wv_b + wo, E, nullptr, vtbuf, nullptr, M, E, E);
    attn_kernel<<<gAttn, 32, 0, stream>>>(qbuf, kbuf, vtbuf, xcat, T, M);
    // FFN: bias + ReLU, bf16 out
    gemm_bf16_kernel<true,true,false,false><<<gProj, 128, 0, stream>>>(
        xcat, E, Wff_b + wo, E, nullptr, xb, bff + (size_t)l * E, M, E, E);
  }
  // logits: fp32 out + bias, N=50257 (store-guarded; loads cover padded Vp)
  const dim3 gOut(Vp / 128, M / 16);           // 393 x 256
  gemm_bf16_kernel<true,false,true,false><<<gOut, 128, 0, stream>>>(
      xb, E, Wout_b, Vp, (float*)d_out, nullptr, bout, M, V, E);
}
